// SparseMoE_77506979824192
// MI455X (gfx1250) — matrix operand order repaired and checked by reading, not compiled, not run on hardware
//
#include <hip/hip_runtime.h>
#include <stdint.h>

// Problem constants (match reference).
#define N_TOK 8192
#define DDIM  512
#define HDIM  2048
#define NEXP  8

typedef __attribute__((ext_vector_type(16))) __bf16 v16bf;
typedef __attribute__((ext_vector_type(8)))  float  v8f;
typedef __attribute__((ext_vector_type(4)))  unsigned int u32x4;
typedef __attribute__((ext_vector_type(8)))  int  i32x8;
typedef __attribute__((ext_vector_type(4)))  int  i32x4;

// POD 16B / 8B vectors (avoid HIP vector-class-in-union issues).
struct __align__(16) U4 { unsigned int x, y, z, w; };
struct __align__(8)  U2 { unsigned int x, y; };

union FragU { U4 q[2]; v16bf v; };

__device__ __forceinline__ unsigned short f32_to_bf16(float f) {
  union { float f; unsigned int u; } c; c.f = f;
  unsigned int u = c.u;
  unsigned int r = u + 0x7FFFu + ((u >> 16) & 1u);   // round-to-nearest-even
  return (unsigned short)(r >> 16);
}

__device__ __forceinline__ v8f wmma_bf16(v16bf a, v16bf b, v8f c) {
  // D = A(16x32 bf16) * B(32x16 bf16) + C(16x16 f32)
  return __builtin_amdgcn_wmma_f32_16x16x32_bf16(false, a, false, b, (short)0, c,
                                                 false, false);
}

// A fragment: matrix row-major in LDS (stride elements). ISA 16-bit A layout:
// lanes 0-15: M=lane, K = {0..7, 16..23}; lanes 16-31: M=lane-16, K = {8..15, 24..31}.
__device__ __forceinline__ v16bf frag_a_lds(const unsigned short* base, int stride,
                                            int row0, int kbase, int lane) {
  int r = row0 + (lane & 15);
  int half = lane >> 4;
  const unsigned short* p = base + r * stride + kbase + half * 8;
  FragU u;
  u.q[0] = *(const U4*)(p);        // K run kbase+half*8 .. +7
  u.q[1] = *(const U4*)(p + 16);   // K run kbase+16+half*8 .. +7
  return u.v;
}

// B fragment: tile stored in LDS as [n][k] (stride elements, TDM-padded rows).
// ISA 16-bit B layout: lane holds column n=lane%16; K = {0..7, 16..23} (+8 for lanes 16-31).
__device__ __forceinline__ v16bf frag_b_lds(const unsigned short* base, int stride,
                                            int col0, int lane) {
  int n = col0 + (lane & 15);
  int kb = (lane >> 4) * 8;
  const unsigned short* p = base + n * stride + kb;
  FragU u;
  u.q[0] = *(const U4*)(p);
  u.q[1] = *(const U4*)(p + 16);
  return u.v;
}

// ---------------------------------------------------------------------------
// Tensor Data Mover: load a dense 2-D bf16 tile [tile_d1 rows x tile_d0 cols]
// from a row-major tensor (row pitch stride0 elements) into LDS at lds_off.
// D# pad fields insert 16B of LDS padding after every 64B row chunk, so a
// 32-element (64B) row lands with an 80B pitch == 40-element frag_b stride.
// Descriptor packing per CDNA5 ISA section 8.3 / 8.4 (groups 2,3 zero: 2-D).
__device__ __forceinline__ void tdm_load_tile_2d(unsigned lds_off,
                                                 const unsigned short* gsrc,
                                                 unsigned tile_d0, unsigned tile_d1,
                                                 unsigned tensor_d0, unsigned tensor_d1,
                                                 unsigned stride0) {
  unsigned long long ga = (unsigned long long)(uintptr_t)gsrc;
  u32x4 g0;
  g0[0] = 1u;                                        // count=1, user descriptor
  g0[1] = lds_off;                                   // LDS byte address
  g0[2] = (unsigned)(ga & 0xFFFFFFFFu);              // global_addr[31:0]
  g0[3] = (unsigned)((ga >> 32) & 0x01FFFFFFu)       // global_addr[56:32]
          | (2u << 30);                              // type = 2 ("image")
  i32x8 g1;
  g1[0] = (int)((1u << 16)                           // data_size = 1 -> 2 bytes
                | (1u << 20)                         // pad_enable
                | (3u << 22)                         // pad_interval: 16 DWORDs (64B)
                | (3u << 25));                       // pad_amount: 4 DWORDs (16B)
  g1[1] = (int)((tensor_d0 & 0xFFFFu) << 16);        // tensor_dim0[15:0]
  g1[2] = (int)((tensor_d0 >> 16) | ((tensor_d1 & 0xFFFFu) << 16));
  g1[3] = (int)((tensor_d1 >> 16) | (tile_d0 << 16));
  g1[4] = (int)tile_d1;                              // tile_dim2 = 0
  g1[5] = (int)stride0;                              // tensor_dim0_stride[31:0]
  g1[6] = 0;                                         // stride0 hi, stride1 lo (unused, 2-D)
  g1[7] = 0;
  i32x4 z4 = (i32x4)0;
#if __has_include(<hip/amd_detail/amd_gfx1250_TDM.h>)
  i32x8 z8 = (i32x8)0;                               // amdgpu-toolchain: 6-arg form
  __builtin_amdgcn_tensor_load_to_lds(g0, g1, z4, z4, z8, 0);
#else
  __builtin_amdgcn_tensor_load_to_lds(g0, g1, z4, z4, 0);  // ROCm 7.2: 5-arg form
#endif
}

// ---------------------------------------------------------------------------
// Kernel 1: zero output accumulator + per-expert counters (every launch).
__global__ void zero_init_kernel(float* __restrict__ out, long long n,
                                 int* __restrict__ cnt) {
  long long i = (long long)blockIdx.x * blockDim.x + threadIdx.x;
  long long stride = (long long)gridDim.x * blockDim.x;
  for (long long j = i; j < n; j += stride) out[j] = 0.0f;
  if (i < NEXP) cnt[i] = 0;
}

// Kernel 2a: f32 -> bf16 conversion (grid-stride), row-major preserved (z).
__global__ void cvt_bf16_kernel(const float* __restrict__ src,
                                unsigned short* __restrict__ dst, long long n) {
  long long i = (long long)blockIdx.x * blockDim.x + threadIdx.x;
  long long stride = (long long)gridDim.x * blockDim.x;
  for (long long j = i; j < n; j += stride) dst[j] = f32_to_bf16(src[j]);
}

// Kernel 2b: per-expert f32 [R][C] -> bf16 transposed [C][R], 32x32 LDS tiles.
// Makes the GEMM K-dimension contiguous so B tiles are dense 2-D TDM tiles.
__global__ __launch_bounds__(256) void cvt_transpose_kernel(
    const float* __restrict__ src, unsigned short* __restrict__ dst,
    int R, int C) {
  __shared__ unsigned short tile[32][33];
  const size_t slab = (size_t)blockIdx.z * R * C;
  const int c0 = blockIdx.x * 32, r0 = blockIdx.y * 32;
  const int tx = threadIdx.x & 31, ty = threadIdx.x >> 5;   // 8 rows per pass
#pragma unroll
  for (int i = ty; i < 32; i += 8)
    tile[i][tx] = f32_to_bf16(src[slab + (size_t)(r0 + i) * C + c0 + tx]);
  __syncthreads();
#pragma unroll
  for (int i = ty; i < 32; i += 8)
    dst[slab + (size_t)(c0 + i) * R + r0 + tx] = tile[tx][i];
}

// Kernel 3: gating. One wave32 per token: logits, softmax over 8, top-2,
// append (token, prob) into per-expert buckets.
__global__ __launch_bounds__(256) void gate_topk_kernel(
    const float* __restrict__ z, const float* __restrict__ gw,
    const float* __restrict__ gb, int* __restrict__ cnt,
    int* __restrict__ tokid, float* __restrict__ tokw) {
  const int wid = threadIdx.x >> 5;
  const int lane = threadIdx.x & 31;
  const int token = blockIdx.x * 8 + wid;
  const float* zr = z + (size_t)token * DDIM;

  float acc[NEXP];
#pragma unroll
  for (int e = 0; e < NEXP; ++e) acc[e] = 0.0f;

  for (int it = 0; it < DDIM / 32; ++it) {
    int d = lane + it * 32;
    float zv = zr[d];
    const float* g = gw + (size_t)d * NEXP;
#pragma unroll
    for (int e = 0; e < NEXP; ++e) acc[e] += zv * g[e];
  }
#pragma unroll
  for (int e = 0; e < NEXP; ++e) {
#pragma unroll
    for (int off = 16; off > 0; off >>= 1)
      acc[e] += __shfl_xor(acc[e], off, 32);
  }

  if (lane == 0) {
    float p[NEXP];
    float m = -3.4e38f;
#pragma unroll
    for (int e = 0; e < NEXP; ++e) {
      p[e] = acc[e] + gb[e];
      m = p[e] > m ? p[e] : m;
    }
    float s = 0.0f;
#pragma unroll
    for (int e = 0; e < NEXP; ++e) { p[e] = __expf(p[e] - m); s += p[e]; }
    float inv = 1.0f / s;
#pragma unroll
    for (int e = 0; e < NEXP; ++e) p[e] *= inv;

    int i1 = 0; float p1 = p[0];
#pragma unroll
    for (int e = 1; e < NEXP; ++e) if (p[e] > p1) { p1 = p[e]; i1 = e; }
    int i2 = -1; float p2 = -1.0f;
#pragma unroll
    for (int e = 0; e < NEXP; ++e)
      if (e != i1 && p[e] > p2) { p2 = p[e]; i2 = e; }

    int pos1 = atomicAdd(&cnt[i1], 1);
    tokid[i1 * N_TOK + pos1] = token;
    tokw[i1 * N_TOK + pos1] = p1;
    int pos2 = atomicAdd(&cnt[i2], 1);
    tokid[i2 * N_TOK + pos2] = token;
    tokw[i2 * N_TOK + pos2] = p2;
  }
}

// Kernel 4: per-expert fused FFN: out[tok] += w * (relu(z@W1+b1)@W2 + b2).
// Block = 32 gathered tokens of one expert, 8 waves. B tiles arrive via the
// Tensor Data Mover (wave 0 issues, TENSORcnt + barrier publishes to all).
__global__ __launch_bounds__(256) void moe_expert_kernel(
    const unsigned short* __restrict__ zb, const unsigned short* __restrict__ w1t,
    const unsigned short* __restrict__ w2t, const float* __restrict__ b1,
    const float* __restrict__ b2, const int* __restrict__ cnt,
    const int* __restrict__ tokid, const float* __restrict__ tokw,
    float* __restrict__ out) {
  const int e  = blockIdx.y;
  const int m0 = blockIdx.x * 32;
  const int ce = cnt[e];
  if (m0 >= ce) return;

  // Padded pitches (x40, x264 elements) keep every fragment load 16B-aligned.
  __shared__ __align__(16) unsigned short aA[32 * 40];    // A chunk 32x32
  __shared__ __align__(16) unsigned short bB[512 * 40];   // TDM B tile [n][k], 80B pitch
  __shared__ __align__(16) unsigned short hsb[32 * 264];  // h strip 32x256 (A of GEMM2)
  __shared__ int   stok[32];
  __shared__ float swgt[32];

  const int tid  = threadIdx.x;
  const int wave = tid >> 5;
  const int lane = tid & 31;

  if (tid < 32) {
    int s = m0 + tid;
    int ok = s < ce;
    stok[tid] = ok ? tokid[e * N_TOK + s] : 0;
    swgt[tid] = ok ? tokw[e * N_TOK + s] : 0.0f;   // pad rows contribute 0
  }
  __syncthreads();

  const unsigned short* w1e = w1t + (size_t)e * HDIM * DDIM;  // [H][D] row-major (pre-T)
  const unsigned short* w2e = w2t + (size_t)e * DDIM * HDIM;  // [D][H] row-major (pre-T)
  const unsigned bB_off = (unsigned)(uintptr_t)(void*)&bB[0]; // LDS byte address

  const int ar = tid >> 3;         // 0..31: A row staged by this thread
  const int ac = (tid & 7) * 4;    // A col quad (4 bf16 = 8B)

  v8f zf = {};
  v8f acc2[2][4];                  // wave owns 32 rows x 64 cols of the 32x512 output
#pragma unroll
  for (int mt = 0; mt < 2; ++mt)
#pragma unroll
    for (int nt = 0; nt < 4; ++nt) acc2[mt][nt] = zf;

  for (int hs = 0; hs < HDIM; hs += 256) {      // 8 strips of H
    // ---- GEMM1: hstrip[32][256] = relu(z[32,512] @ W1[:,hs:hs+256] + b1) ----
    v8f acc1[2][2];                // wave owns 32 rows x 32 cols of the strip
#pragma unroll
    for (int mt = 0; mt < 2; ++mt)
#pragma unroll
      for (int nt = 0; nt < 2; ++nt) acc1[mt][nt] = zf;

    for (int kc = 0; kc < DDIM; kc += 32) {
      __syncthreads();                           // prior tile reads complete
      {  // stage A: gathered token rows, 32x32 bf16 (tiny, plain copy)
        const unsigned short* zp = zb + (size_t)stok[ar] * DDIM + kc + ac;
        __builtin_prefetch(zp + 32, 0, 1);
        *(U2*)&aA[ar * 40 + ac] = *(const U2*)zp;
      }
      if (wave == 0) {  // B1 tile: 256 N-rows x 32 K from w1^T[e] ([H][D])
        tdm_load_tile_2d(bB_off, w1e + (size_t)hs * DDIM + kc,
                         /*tile_d0=*/32, /*tile_d1=*/256,
                         /*tensor_d0=*/DDIM, /*tensor_d1=*/HDIM,
                         /*stride0=*/DDIM);
      }
      __builtin_amdgcn_s_wait_tensorcnt(0);      // wave0 waits on TDM; others no-op
      __syncthreads();                           // publish aA + bB to all waves

      v16bf af[2], bf[2];
#pragma unroll
      for (int mt = 0; mt < 2; ++mt) af[mt] = frag_a_lds(aA, 40, mt * 16, 0, lane);
#pragma unroll
      for (int nt = 0; nt < 2; ++nt) bf[nt] = frag_b_lds(bB, 40, wave * 32 + nt * 16, lane);
#pragma unroll
      for (int mt = 0; mt < 2; ++mt)
#pragma unroll
        for (int nt = 0; nt < 2; ++nt)
          acc1[mt][nt] = wmma_bf16(af[mt], bf[nt], acc1[mt][nt]);
    }

    // bias + relu + bf16; wave writes its own 32-col slice of the strip
#pragma unroll
    for (int mt = 0; mt < 2; ++mt)
#pragma unroll
      for (int nt = 0; nt < 2; ++nt) {
        int col = wave * 32 + nt * 16 + (lane & 15);
        float bv = b1[(size_t)e * HDIM + hs + col];
#pragma unroll
        for (int v = 0; v < 8; ++v) {
          int lr = mt * 16 + (lane >> 4) * 8 + v;   // C layout: M = v + 8*(lane>>4)
          float h = acc1[mt][nt][v] + bv;
          h = h > 0.0f ? h : 0.0f;
          hsb[lr * 264 + col] = f32_to_bf16(h);
        }
      }

    // ---- GEMM2 partial: acc2 += hstrip[32,256] @ W2[hs:hs+256, 0:512] ----
    for (int kc = 0; kc < 256; kc += 32) {
      __syncthreads();                           // hsb published / bB reads done
      if (wave == 0) {  // B2 tile: 512 N-rows x 32 K from w2^T[e] ([D][H])
        tdm_load_tile_2d(bB_off, w2e + (size_t)(hs + kc),
                         /*tile_d0=*/32, /*tile_d1=*/512,
                         /*tensor_d0=*/HDIM, /*tensor_d1=*/DDIM,
                         /*stride0=*/HDIM);
      }
      __builtin_amdgcn_s_wait_tensorcnt(0);
      __syncthreads();

      v16bf af[2], bf2[4];
#pragma unroll
      for (int mt = 0; mt < 2; ++mt) af[mt] = frag_a_lds(hsb, 264, mt * 16, kc, lane);
#pragma unroll
      for (int nt = 0; nt < 4; ++nt) bf2[nt] = frag_b_lds(bB, 40, wave * 64 + nt * 16, lane);
#pragma unroll
      for (int mt = 0; mt < 2; ++mt)
#pragma unroll
        for (int nt = 0; nt < 4; ++nt)
          acc2[mt][nt] = wmma_bf16(af[mt], bf2[nt], acc2[mt][nt]);
    }
  }

  // Epilogue: out[token] += w * (acc2 + b2). Pad rows have w == 0.
#pragma unroll
  for (int mt = 0; mt < 2; ++mt)
#pragma unroll
    for (int nt = 0; nt < 4; ++nt) {
      int col = wave * 64 + nt * 16 + (lane & 15);
      float b2v = b2[(size_t)e * DDIM + col];
#pragma unroll
      for (int v = 0; v < 8; ++v) {
        int lr = mt * 16 + (lane >> 4) * 8 + v;
        float w = swgt[lr];
        int t = stok[lr];
        atomicAdd(out + (size_t)t * DDIM + col, w * (acc2[mt][nt][v] + b2v));
      }
    }
}

// ---------------------------------------------------------------------------
extern "C" void kernel_launch(void* const* d_in, const int* in_sizes, int n_in,
                              void* d_out, int out_size, void* d_ws, size_t ws_size,
                              hipStream_t stream) {
  const float* z  = (const float*)d_in[0];
  const float* gw = (const float*)d_in[1];
  const float* gb = (const float*)d_in[2];
  const float* w1 = (const float*)d_in[3];
  const float* b1 = (const float*)d_in[4];
  const float* w2 = (const float*)d_in[5];
  const float* b2 = (const float*)d_in[6];
  // d_in[7] is k (== 2), hardcoded.
  float* out = (float*)d_out;

  // Workspace layout (~42.5 MB):
  char* ws = (char*)d_ws;
  unsigned short* zb  = (unsigned short*)ws;                       //  8 MB, z bf16
  unsigned short* w1t = zb  + (size_t)N_TOK * DDIM;                // 16 MB, [E][H][D]
  unsigned short* w2t = w1t + (size_t)NEXP * DDIM * HDIM;          // 16 MB, [E][D][H]
  int*   cnt   = (int*)(w2t + (size_t)NEXP * HDIM * DDIM);         // 8 ints (+pad)
  int*   tokid = cnt + 64;                                         // 256 KB
  float* tokw  = (float*)(tokid + NEXP * N_TOK);                   // 256 KB
  (void)in_sizes; (void)n_in; (void)out_size; (void)ws_size;

  zero_init_kernel<<<1024, 256, 0, stream>>>(out, (long long)N_TOK * DDIM, cnt);
  cvt_bf16_kernel<<<2048, 256, 0, stream>>>(z, zb, (long long)N_TOK * DDIM);
  // w1 [E][D][H] -> w1t [E][H][D];  w2 [E][H][D'] -> w2t [E][D'][H]
  cvt_transpose_kernel<<<dim3(HDIM / 32, DDIM / 32, NEXP), 256, 0, stream>>>(
      w1, w1t, DDIM, HDIM);
  cvt_transpose_kernel<<<dim3(DDIM / 32, HDIM / 32, NEXP), 256, 0, stream>>>(
      w2, w2t, HDIM, DDIM);
  gate_topk_kernel<<<N_TOK / 8, 256, 0, stream>>>(z, gw, gb, cnt, tokid, tokw);
  moe_expert_kernel<<<dim3(N_TOK / 32, NEXP), 256, 0, stream>>>(
      zb, w1t, w2t, b1, b2, cnt, tokid, tokw, out);
}
